// GAT_Decoder_54597624266991
// MI455X (gfx1250) — matrix operand order repaired and checked by reading, not compiled, not run on hardware
//
#include <hip/hip_runtime.h>
#include <cstddef>

// ---------------------------------------------------------------------------
// Problem constants (GAT decoder): B=256, N=1000, D=512, H=8, HD=64
// ---------------------------------------------------------------------------
#define BB 256
#define NN 1000
#define DD 512
#define HH 8
#define HDD 64
#define NORM_MHA 0.125f                 // 1/sqrt(64)
#define NORM_PTR 0.04419417382415922f   // 1/sqrt(512)

typedef __attribute__((ext_vector_type(16))) __bf16          v16bf;
typedef __attribute__((ext_vector_type(2)))  __bf16          v2bf;
typedef __attribute__((ext_vector_type(8)))  float           v8f;
typedef __attribute__((ext_vector_type(16))) unsigned short  v16us;

// hardware f32 -> bf16 convert (lowers to v_cvt_pk_bf16_f32 on gfx1250)
__device__ __forceinline__ unsigned short f2bf(float f) {
    __bf16 h = (__bf16)f;
    return __builtin_bit_cast(unsigned short, h);
}
// pack two f32 into two bf16 in one u32 (lo -> low 16 bits)
__device__ __forceinline__ unsigned int pk2bf(float lo, float hi) {
    v2bf v; v[0] = (__bf16)lo; v[1] = (__bf16)hi;
    return __builtin_bit_cast(unsigned int, v);
}

// ---------------------------------------------------------------------------
// Kernel 1: state[b,d] = dec_in[b] @ W_fc + pool[b] @ W_fc1   (B x D)
// ---------------------------------------------------------------------------
__global__ void state_kernel(const float* __restrict__ enc,
                             const float* __restrict__ pool,
                             const float* __restrict__ dyn,
                             const float* __restrict__ Wfc,
                             const float* __restrict__ Wfc1,
                             float* __restrict__ state) {
    int gid = blockIdx.x * 256 + threadIdx.x;     // B*D threads
    int b = gid >> 9, d = gid & 511;
    const float* e0 = enc + (size_t)b * NN * DD;  // encoder_inputs[b,0,:]
    const float* pr = pool + (size_t)b * DD;
    float s = dyn[b] * Wfc[(size_t)DD * DD + d];  // last row of (D+1,D)
    for (int k = 0; k < DD; ++k)
        s += e0[k] * Wfc[(size_t)k * DD + d] + pr[k] * Wfc1[(size_t)k * DD + d];
    state[gid] = s;
}

// ---------------------------------------------------------------------------
// Kernel 2: Q = state @ Wq  (B x D, fp32)
// ---------------------------------------------------------------------------
__global__ void q_kernel(const float* __restrict__ state,
                         const float* __restrict__ Wq,
                         float* __restrict__ Qm) {
    int gid = blockIdx.x * 256 + threadIdx.x;
    int b = gid >> 9, d = gid & 511;
    const float* sr = state + (size_t)b * DD;
    float s = 0.f;
    for (int k = 0; k < DD; ++k) s += sr[k] * Wq[(size_t)k * DD + d];
    Qm[gid] = s;
}

// ---------------------------------------------------------------------------
// Kernel 3: Wcomb[b,k,h] = sum_d' Wk_mha[k, h*64+d'] * Q[b, h*64+d'],
// stored as pre-swizzled bf16 WMMA B-fragments:
//   layout [b][kstep t (16)][lane (32)][elem (16)], elem e of lane l = row
//   t*32 + (l>>4)*16 + e, col = l&15 (cols 8..15 zero-padded).
// ---------------------------------------------------------------------------
__global__ void wcomb_kernel(const float* __restrict__ Wk,
                             const float* __restrict__ Qm,
                             unsigned short* __restrict__ outw) {
    int gid = blockIdx.x * 256 + threadIdx.x;   // B*16*32 threads
    int b = gid >> 9;
    int rem = gid & 511;
    int t = rem >> 5;
    int l = rem & 31;
    int col = l & 15;
    int lh = l >> 4;
    v16us vals;
    if (col < HH) {
        const float* qh = Qm + (size_t)b * DD + col * HDD;
        #pragma unroll
        for (int e = 0; e < 16; ++e) {
            int krow = t * 32 + lh * 16 + e;
            const float* wr = Wk + (size_t)krow * DD + col * HDD;
            float s = 0.f;
            for (int d = 0; d < HDD; ++d) s += wr[d] * qh[d];
            vals[e] = f2bf(s);
        }
    } else {
        #pragma unroll
        for (int e = 0; e < 16; ++e) vals[e] = 0;
    }
    *(v16us*)(outw + (size_t)gid * 16) = vals;
}

// ---------------------------------------------------------------------------
// Kernel 4: fully fused per-batch attention + pointer softmax.
// One block (8 waves, 256 threads) per batch b. enc[b] (2MB) stays L2-hot
// across the three passes; device-level traffic ~= one read of enc (524MB).
// ---------------------------------------------------------------------------
__global__ void __launch_bounds__(256)
attn_kernel(const float* __restrict__ enc,
            const int*   __restrict__ mask,
            const unsigned short* __restrict__ wcomb,
            const float* __restrict__ Wv,
            const float* __restrict__ Wo,
            const float* __restrict__ Wkp,
            float* __restrict__ out) {
    __shared__ __align__(16) unsigned char smem[65536];
    // region map (reused across stages):
    float* compat = (float*)smem;                       // [1024][8] f32, 32KB (A,B,C)
    unsigned short* tileT = (unsigned short*)(smem + 32768); // [512][32] bf16, 32KB (C)
    float* t_lds   = (float*)smem;                      // [8][512] f32, 16KB (D)
    float* mha_pre = (float*)(smem + 16384);            // 512 f32
    float* mha_o   = (float*)(smem + 18432);            // 512 f32
    float* w2      = (float*)(smem + 20480);            // 512 f32
    float* c2      = (float*)(smem + 22528);            // 1024 f32
    float* red     = (float*)(smem + 26624);            // reduce scratch

    const int b    = blockIdx.x;
    const int tid  = threadIdx.x;
    const int wid  = tid >> 5;
    const int lane = tid & 31;
    const int lh   = lane >> 4;   // lane half
    const int lm   = lane & 15;
    const int* mrow = mask + (size_t)b * NN;
    const float* encb = enc + (size_t)b * NN * DD;

    // ---------- Stage A: compat[n,h] = norm * (enc[b] @ Wcomb[b]) via WMMA ----
    // each wave owns 16-row n-tiles, stride 128 rows across 8 waves.
    // Out-of-range rows are address-clamped (not predicated): a lane's A-row
    // only feeds output row M=lane&15, which is discarded at the compat write
    // when n0+M >= N, so clamped-row garbage never escapes.
    for (int n0 = wid * 16; n0 < NN; n0 += 128) {
        int nrow = n0 + lm;                 // this lane's A row (both halves)
        int nclamp = nrow < NN ? nrow : NN - 1;
        const float* arow = encb + (size_t)nclamp * DD;
        if (nrow + 128 < NN) __builtin_prefetch(encb + (size_t)(nrow + 128) * DD, 0, 0);
        v8f acc = {0.f,0.f,0.f,0.f,0.f,0.f,0.f,0.f};
        const v16us* bfr = (const v16us*)(wcomb + ((((size_t)b * 16) * 32 + lane) * 16));
        #pragma unroll 4
        for (int t = 0; t < 16; ++t) {
            int kk = t * 32 + lh * 16;      // A layout: contiguous run [kk,kk+16)
            float4 f0 = *(const float4*)(arow + kk);
            float4 f1 = *(const float4*)(arow + kk + 4);
            float4 f2 = *(const float4*)(arow + kk + 8);
            float4 f3 = *(const float4*)(arow + kk + 12);
            v16bf av;
            av[0]=(__bf16)f0.x; av[1]=(__bf16)f0.y; av[2]=(__bf16)f0.z; av[3]=(__bf16)f0.w;
            av[4]=(__bf16)f1.x; av[5]=(__bf16)f1.y; av[6]=(__bf16)f1.z; av[7]=(__bf16)f1.w;
            av[8]=(__bf16)f2.x; av[9]=(__bf16)f2.y; av[10]=(__bf16)f2.z; av[11]=(__bf16)f2.w;
            av[12]=(__bf16)f3.x; av[13]=(__bf16)f3.y; av[14]=(__bf16)f3.z; av[15]=(__bf16)f3.w;
            v16us bu = bfr[(size_t)t * 32];   // pre-swizzled fragment, 32B/lane
            acc = __builtin_amdgcn_wmma_f32_16x16x32_bf16(
                false, av, false, __builtin_bit_cast(v16bf, bu),
                (short)0, acc, false, false);
        }
        if (lm < HH) {
            #pragma unroll
            for (int r = 0; r < 8; ++r) {
                int n = n0 + r + lh * 8;      // C layout: M = r + 8*(lane>=16)
                if (n < NN) compat[n * HH + lm] = acc[r] * NORM_MHA;
            }
        }
    }
    __syncthreads();

    // ---------- Stage B: masked softmax per head (wave w -> head w) ----------
    {
        int h = wid;
        float mx = -INFINITY;
        for (int n = lane; n < NN; n += 32) {
            float u = mrow[n] ? -INFINITY : compat[n * HH + h];
            mx = fmaxf(mx, u);
        }
        #pragma unroll
        for (int off = 16; off; off >>= 1) mx = fmaxf(mx, __shfl_xor(mx, off, 32));
        float s = 0.f;
        for (int n = lane; n < NN; n += 32) {
            float u = mrow[n] ? -INFINITY : compat[n * HH + h];
            s += (u == -INFINITY) ? 0.f : __expf(u - mx);
        }
        #pragma unroll
        for (int off = 16; off; off >>= 1) s += __shfl_xor(s, off, 32);
        float inv = (s > 0.f) ? 1.f / s : 0.f;
        for (int n = lane; n < 1024; n += 32) {     // also zero-pad rows 1000..1023
            float a = 0.f;
            if (n < NN) {
                float u = mrow[n] ? -INFINITY : compat[n * HH + h];
                a = (u == -INFINITY) ? 0.f : __expf(u - mx) * inv;
            }
            compat[n * HH + h] = a;                 // in-place attn
        }
    }
    __syncthreads();

    // ---------- Stage C: t[h,:] = attn[h,:] @ enc[b] via WMMA ----------------
    // wave w owns cols [w*64, w*64+64); K (=n) staged 32 rows/step into LDS.
    // Rows k >= N are address-clamped: they only multiply attn rows that were
    // zero-padded in stage B, so their (finite) garbage contributes 0.
    {
        int c0 = wid * 64;
        v8f tac[4];
        #pragma unroll
        for (int ct = 0; ct < 4; ++ct)
            tac[ct] = (v8f){0.f,0.f,0.f,0.f,0.f,0.f,0.f,0.f};
        // staging map: thread owns row-pair (2rp, 2rp+1) x 32 cols; packed u32 stores
        const int rp = tid >> 4;            // 0..15
        const int cs = (tid & 15) * 32;     // col segment
        for (int t = 0; t < 32; ++t) {
            int k0 = t * 32;
            __syncthreads();     // previous-step fragments consumed
            {
                int n0r = k0 + 2 * rp;
                int r0 = n0r < NN ? n0r : NN - 1;
                int r1 = (n0r + 1) < NN ? (n0r + 1) : NN - 1;
                const float* s0 = encb + (size_t)r0 * DD + cs;
                const float* s1 = encb + (size_t)r1 * DD + cs;
                #pragma unroll
                for (int i = 0; i < 32; i += 4) {
                    float4 fa = *(const float4*)(s0 + i);
                    float4 fb = *(const float4*)(s1 + i);
                    *(unsigned int*)(tileT + (cs + i + 0) * 32 + 2 * rp) = pk2bf(fa.x, fb.x);
                    *(unsigned int*)(tileT + (cs + i + 1) * 32 + 2 * rp) = pk2bf(fa.y, fb.y);
                    *(unsigned int*)(tileT + (cs + i + 2) * 32 + 2 * rp) = pk2bf(fa.z, fb.z);
                    *(unsigned int*)(tileT + (cs + i + 3) * 32 + 2 * rp) = pk2bf(fa.w, fb.w);
                }
            }
            __syncthreads();
            // A fragment from attn (rows = heads, K = n); rows 8..15 zero
            v16bf av;
            if (lm < HH) {
                #pragma unroll
                for (int e = 0; e < 16; ++e) {
                    int K = e + lh * 16;
                    av[e] = (__bf16)compat[(k0 + K) * HH + lm];
                }
            } else {
                #pragma unroll
                for (int e = 0; e < 16; ++e) av[e] = (__bf16)0.f;
            }
            #pragma unroll
            for (int ct = 0; ct < 4; ++ct) {
                int col = c0 + ct * 16 + lm;
                v16us bu = *(const v16us*)(tileT + col * 32 + lh * 16); // 32B contig
                tac[ct] = __builtin_amdgcn_wmma_f32_16x16x32_bf16(
                    false, av, false, __builtin_bit_cast(v16bf, bu),
                    (short)0, tac[ct], false, false);
            }
        }
        __syncthreads();         // attn fully consumed; repurpose region as t_lds
        if (lane < 16) {         // lanes<16 hold M=r (valid heads 0..7)
            #pragma unroll
            for (int ct = 0; ct < 4; ++ct)
                #pragma unroll
                for (int r = 0; r < 8; ++r)
                    t_lds[r * DD + c0 + ct * 16 + lm] = tac[ct][r];
        }
    }
    __syncthreads();

    // ---------- Stage D: mha_pre[c] = t[h(c),:] @ Wv[:,c]; then @ Wo ---------
    for (int c = tid; c < DD; c += 256) {
        const float* th = t_lds + (c >> 6) * DD;
        float s = 0.f;
        for (int k = 0; k < DD; ++k) s += th[k] * Wv[(size_t)k * DD + c];
        mha_pre[c] = s;
    }
    __syncthreads();
    for (int d = tid; d < DD; d += 256) {
        float s = 0.f;
        for (int k = 0; k < DD; ++k) s += mha_pre[k] * Wo[(size_t)k * DD + d];
        mha_o[d] = s;
    }
    __syncthreads();

    // ---------- Stage E: w2 = Wk_ptr @ mha_out; compat2[n] = enc[n]·w2 -------
    for (int k = tid; k < DD; k += 256) {
        const float* wr = Wkp + (size_t)k * DD;
        float s = 0.f;
        for (int d = 0; d < DD; ++d) s += wr[d] * mha_o[d];
        w2[k] = s;
    }
    __syncthreads();
    for (int n = wid; n < NN; n += 8) {
        const float* er = encb + (size_t)n * DD;
        float s = 0.f;
        for (int k = lane; k < DD; k += 32) s += er[k] * w2[k];
        #pragma unroll
        for (int off = 16; off; off >>= 1) s += __shfl_xor(s, off, 32);
        if (lane == 0) {
            float x = tanhf(s * NORM_PTR) * 10.0f;
            c2[n] = mrow[n] ? -INFINITY : x;
        }
    }
    __syncthreads();

    // ---------- Stage F: final softmax over n, write scores ------------------
    {
        float mx = -INFINITY;
        for (int n = tid; n < NN; n += 256) mx = fmaxf(mx, c2[n]);
        #pragma unroll
        for (int off = 16; off; off >>= 1) mx = fmaxf(mx, __shfl_xor(mx, off, 32));
        if (lane == 0) red[wid] = mx;
        __syncthreads();
        float m = -INFINITY;
        #pragma unroll
        for (int i = 0; i < 8; ++i) m = fmaxf(m, red[i]);
        __syncthreads();
        float s = 0.f;
        for (int n = tid; n < NN; n += 256) {
            float x = c2[n];
            s += (x == -INFINITY) ? 0.f : __expf(x - m);
        }
        #pragma unroll
        for (int off = 16; off; off >>= 1) s += __shfl_xor(s, off, 32);
        if (lane == 0) red[8 + wid] = s;
        __syncthreads();
        float sum = 0.f;
        #pragma unroll
        for (int i = 0; i < 8; ++i) sum += red[8 + i];
        bool allm = (m == -INFINITY);           // all_masked fallback -> uniform
        float inv = allm ? 0.f : 1.f / sum;
        float* orow = out + (size_t)b * NN;
        for (int n = tid; n < NN; n += 256) {
            float x = c2[n];
            orow[n] = allm ? (1.f / NN)
                           : ((x == -INFINITY) ? 0.f : __expf(x - m) * inv);
        }
    }
}

// ---------------------------------------------------------------------------
extern "C" void kernel_launch(void* const* d_in, const int* in_sizes, int n_in,
                              void* d_out, int out_size, void* d_ws, size_t ws_size,
                              hipStream_t stream) {
    (void)in_sizes; (void)n_in; (void)out_size; (void)ws_size;
    const float* enc  = (const float*)d_in[0];
    const float* pool = (const float*)d_in[1];
    const float* dyn  = (const float*)d_in[2];
    const int*   mask = (const int*)d_in[3];
    const float* Wfc  = (const float*)d_in[4];
    const float* Wfc1 = (const float*)d_in[5];
    const float* Wq   = (const float*)d_in[6];
    const float* Wk   = (const float*)d_in[7];
    const float* Wv   = (const float*)d_in[8];
    const float* Wo   = (const float*)d_in[9];
    const float* Wkp  = (const float*)d_in[10];
    float* out = (float*)d_out;

    char* ws = (char*)d_ws;
    float* state = (float*)ws;                        // 512 KB
    float* Qm    = (float*)(ws + (1 << 19));          // 512 KB
    unsigned short* wcomb = (unsigned short*)(ws + (1 << 20)); // 4 MB

    state_kernel<<<512, 256, 0, stream>>>(enc, pool, dyn, Wfc, Wfc1, state);
    q_kernel   <<<512, 256, 0, stream>>>(state, Wq, Qm);
    wcomb_kernel<<<512, 256, 0, stream>>>(Wk, Qm, wcomb);
    attn_kernel<<<BB, 256, 0, stream>>>(enc, mask, wcomb, Wv, Wo, Wkp, out);
}
